// SelfAttention_8151847927968
// MI455X (gfx1250) — matrix operand order repaired and checked by reading, not compiled, not run on hardware
//
#include <hip/hip_runtime.h>
#include <math.h>

// Problem constants (from reference)
#define BATCH   8
#define SEQ_T   16
#define DIM     2048
#define NHEAD   16
#define HDIM    128
#define NCACHE  4096
#define NTOK    (BATCH * SEQ_T)      // 128
#define NCHUNK  128                  // full 32-key chunks in cache
#define OUT_ELEMS (NTOK * DIM)       // 262144 per tensor section

typedef __attribute__((ext_vector_type(16))) __bf16 v16bf;
typedef __attribute__((ext_vector_type(8)))  float  v8f;
typedef __attribute__((ext_vector_type(4)))  unsigned int u32x4;
typedef __attribute__((ext_vector_type(8)))  int i32x8;
typedef __attribute__((ext_vector_type(4)))  int i32x4;

#if __has_builtin(__builtin_amdgcn_tensor_load_to_lds) && \
    __has_builtin(__builtin_amdgcn_s_wait_tensorcnt)
#define TDM_OK 1
#else
#define TDM_OK 0
#endif

// LDS row stride for TDM-staged W tiles: 32 data dwords + 4 pad dwords.
// 36 dwords = 144B: rows 16B-aligned (ds_load_b128) and 36*n mod 64 distinct
// for n=0..15 -> bank-conflict-free column reads.
#define WROW 36
#define WTILE_DWORDS (64 * WROW)     // 2304 dwords = 9216 B per 64x32 tile

#if TDM_OK
// DMA a 64-row x 32-float tile of W (row stride DIM floats) into LDS at lds_off.
// D# per CDNA5 ISA ch.8: 2-D tensor, data_size=4B, pad 4 dwords per 32.
__device__ __forceinline__ void tdm_load_w_tile(const float* gsrc, unsigned lds_off) {
  unsigned long long ga = (unsigned long long)(const void*)gsrc;
  u32x4 g0;
  g0[0] = 1u;                                           // count=1, user D#
  g0[1] = lds_off;                                      // lds_addr (bytes)
  g0[2] = (unsigned)(ga & 0xFFFFFFFFu);                 // global_addr[31:0]
  g0[3] = (unsigned)((ga >> 32) & 0x01FFFFFFu)          // global_addr[56:32]
          | (2u << 30);                                 // type = 2 ("image")
  i32x8 g1;
  g1[0] = (2 << 16)                                     // data_size = 4B
        | (1 << 20)                                     // pad_enable
        | (4 << 22)                                     // pad_interval: every 32 dw
        | (3 << 25);                                    // pad_amount: 4 dwords
  g1[1] = (32 << 16);                                   // tensor_dim0 = 32
  g1[2] = (64 << 16);                                   // tensor_dim1 = 64
  g1[3] = (32 << 16);                                   // tile_dim0  = 32
  g1[4] = 64;                                           // tile_dim1=64, tile_dim2=0
  g1[5] = DIM;                                          // tensor_dim0_stride = 2048
  g1[6] = 0;
  g1[7] = 0;
  i32x4 z = {0, 0, 0, 0};
#if defined(__clang_major__) && __clang_major__ >= 23
  i32x8 z8 = {0, 0, 0, 0, 0, 0, 0, 0};
  __builtin_amdgcn_tensor_load_to_lds(g0, g1, z, z, z8, 0);
#else
  __builtin_amdgcn_tensor_load_to_lds(g0, g1, z, z, 0);
#endif
}
#endif

// D = A(16x32 bf16) x B(32x16 bf16) + C(16x16 f32); RA/RB = operand-reuse hints
template <bool RA, bool RB>
__device__ __forceinline__ v8f wmma_bf16(v16bf a, v16bf b, v8f c) {
  return __builtin_amdgcn_wmma_f32_16x16x32_bf16(
      /*neg_a=*/false, a, /*neg_b=*/false, b,
      /*c_mod=*/(short)0, c, /*reuse_a=*/RA, /*reuse_b=*/RB);
}

// ---- fragment loaders (fp32 memory -> bf16 fragments) -----------------------
// A-matrix 16x32 (MxK), row-major source with leading dim ld.
// Lane L(0..15): M=L, K=0..15 in elems[0..15]
// Lane L(16..31): M=L-16, K=16..31.  64B contiguous per lane.
__device__ __forceinline__ v16bf load_A_rm(const float* __restrict__ src, int ld) {
  int lane = threadIdx.x & 31;
  int m  = lane & 15;
  int kb = (lane >> 4) << 4;               // 0 or 16
  const float* p = src + m * ld + kb;
  v16bf a;
#pragma unroll
  for (int i = 0; i < 8; ++i) a[i] = (__bf16)p[i];
#pragma unroll
  for (int i = 0; i < 8; ++i) a[8 + i] = (__bf16)p[8 + i];
  return a;
}

// B-matrix 32x16 (KxN) where B = rows^T: B[k][n] = src[n*ld + k].
// Lane L: N=L&15, K=(L>>4)*16 .. +15 -> 64B contiguous per lane.
__device__ __forceinline__ v16bf load_BT_rm(const float* __restrict__ src, int ld) {
  int lane = threadIdx.x & 31;
  int n  = lane & 15;
  int kb = (lane >> 4) << 4;               // 0 or 16
  const float* p = src + n * ld + kb;
  v16bf b;
#pragma unroll
  for (int i = 0; i < 16; ++i) b[i] = (__bf16)p[i];
  return b;
}

// A-fragment for V^T tile (16 d x 32 keys): A[m=d_local][k=key] = V[k][d0+m].
// Lane index runs along the contiguous d dimension -> wave-coalesced loads.
__device__ __forceinline__ v16bf load_A_vt(const float* __restrict__ src, int d0,
                                           bool clampK) {
  int lane = threadIdx.x & 31;
  int m  = lane & 15;
  int kb = (lane >> 4) << 3;               // 0 or 8
  v16bf a;
#pragma unroll
  for (int i = 0; i < 8; ++i) {
    int k = kb + i;                        // 0..15: always valid
    a[i] = (__bf16)src[k * HDIM + d0 + m];
  }
#pragma unroll
  for (int i = 0; i < 8; ++i) {
    int k = clampK ? 0 : (kb + 16 + i);    // 16..31: clamp on tail chunk
    a[8 + i] = (__bf16)src[k * HDIM + d0 + m];
  }
  return a;
}

#if TDM_OK
// B-fragment from a TDM-staged LDS tile (row stride WROW dwords, 16B aligned):
// lane n reads 16 contiguous floats of row (j*16+n) as 4x ds_load_b128.
__device__ __forceinline__ v16bf load_BT_lds(const float* tile /*AS(3) array elem*/,
                                             int j) {
  int lane = threadIdx.x & 31;
  int n  = lane & 15;
  int kb = (lane >> 4) << 4;               // 0 or 16
  const float4* p = (const float4*)(tile + (j * 16 + n) * WROW + kb);
  v16bf b;
#pragma unroll
  for (int q = 0; q < 4; ++q) {
    float4 f = p[q];
    b[q * 4 + 0] = (__bf16)f.x;
    b[q * 4 + 1] = (__bf16)f.y;
    b[q * 4 + 2] = (__bf16)f.z;
    b[q * 4 + 3] = (__bf16)f.w;
  }
  return b;
}
#endif

// ---- kernel 1: fused QKV projection (TDM-staged weight tiles) ---------------
// q/k/v[m][n] = sum_k x[m][k]*W[n][k] + bias[n]; outputs in [B,H,T,HD] layout.
// 4 waves / block; each wave owns one (matrix, m-tile, 64-col group) task.
__global__ __launch_bounds__(128) void qkv_proj_kernel(
    const float* __restrict__ x,
    const float* __restrict__ Wq, const float* __restrict__ bq,
    const float* __restrict__ Wk, const float* __restrict__ bk,
    const float* __restrict__ Wv, const float* __restrict__ bv,
    float* __restrict__ qws, float* __restrict__ kout, float* __restrict__ vout) {
#if TDM_OK
  __shared__ __align__(16) float wstage[4][2][WTILE_DWORDS];  // 72 KB, dbl-buffered
#endif
  int wave  = threadIdx.x >> 5;
  int lane  = threadIdx.x & 31;
  int wid   = blockIdx.x * 4 + wave;
  int which = wid >> 8;                    // 0=q 1=k 2=v (256 wave-tasks each)
  int t     = wid & 255;
  int m0    = (t >> 5) << 4;               // 8 M-tiles
  int n0    = (t & 31) << 6;               // 32 groups of 4 N-tiles (64 cols)

  const float* W    = (which == 0) ? Wq : (which == 1) ? Wk : Wv;
  const float* bias = (which == 0) ? bq : (which == 1) ? bk : bv;
  float*       dst  = (which == 0) ? qws : (which == 1) ? kout : vout;

  v8f c[4] = {};
#if TDM_OK
  unsigned lds_off[2] = {
      (unsigned)(unsigned long long)(void*)&wstage[wave][0][0],
      (unsigned)(unsigned long long)(void*)&wstage[wave][1][0]};
  // prime buffer 0 with the k0=0 tile
  tdm_load_w_tile(W + (size_t)n0 * DIM, lds_off[0]);
  for (int k0 = 0; k0 < DIM; k0 += 32) {
    int cur = (k0 >> 5) & 1;
    if (k0 + 32 < DIM) {
      tdm_load_w_tile(W + (size_t)n0 * DIM + (k0 + 32), lds_off[cur ^ 1]);
      __builtin_amdgcn_s_wait_tensorcnt(1);   // current tile landed
    } else {
      __builtin_amdgcn_s_wait_tensorcnt(0);
    }
    v16bf a = load_A_rm(x + (size_t)m0 * DIM + k0, DIM);
#pragma unroll
    for (int j = 0; j < 4; ++j) {
      v16bf b = load_BT_lds(&wstage[wave][cur][0], j);
      if (j == 0) c[j] = wmma_bf16<false, false>(a, b, c[j]);
      else        c[j] = wmma_bf16<true,  false>(a, b, c[j]);  // same A as prev
    }
  }
#else
  for (int k0 = 0; k0 < DIM; k0 += 32) {
    v16bf a = load_A_rm(x + (size_t)m0 * DIM + k0, DIM);
#pragma unroll
    for (int j = 0; j < 4; ++j) {
      v16bf b = load_BT_rm(W + (size_t)(n0 + j * 16) * DIM + k0, DIM);
      if (j == 0) c[j] = wmma_bf16<false, false>(a, b, c[j]);
      else        c[j] = wmma_bf16<true,  false>(a, b, c[j]);
    }
  }
#endif

  // C layout: lane -> N=lane&15, M = r + 8*(lane>>4)
  int nn = lane & 15;
  int mh = (lane >> 4) << 3;
#pragma unroll
  for (int j = 0; j < 4; ++j) {
#pragma unroll
    for (int r = 0; r < 8; ++r) {
      int row = m0 + mh + r;               // b*16 + t
      int col = n0 + j * 16 + nn;          // h*128 + hd
      float val = c[j][r] + bias[col];
      int bi = row >> 4, ti = row & 15, hh = col >> 7, hd = col & 127;
      dst[(((size_t)bi * NHEAD + hh) * SEQ_T + ti) * HDIM + hd] = val;
    }
  }
}

// ---- kernel 2: flash attention over 4096 cached + 16 new keys ---------------
__global__ __launch_bounds__(256) void attn_kernel(
    const float* __restrict__ qws, const float* __restrict__ Kc,
    const float* __restrict__ Vc, const float* __restrict__ knew,
    const float* __restrict__ vnew, float* __restrict__ attn_ws) {
  __shared__ __align__(16) __bf16 P_lds[8][SEQ_T * 32];   // 8 KB: P (row-major [q][32])
  __shared__ float O_lds[8][SEQ_T * HDIM];                // 64 KB: wave partials [q][d]
  __shared__ float m_lds[8][SEQ_T];
  __shared__ float l_lds[8][SEQ_T];
  __shared__ float a_bcast[8][SEQ_T];                     // per-chunk alpha broadcast

  int bh   = blockIdx.x;                   // b*16 + h
  int b    = bh >> 4, h = bh & 15;
  int wave = threadIdx.x >> 5;
  int lane = threadIdx.x & 31;

  const float* qb = qws  + (size_t)bh * SEQ_T  * HDIM;
  const float* Kb = Kc   + (size_t)bh * NCACHE * HDIM;
  const float* Vb = Vc   + (size_t)bh * NCACHE * HDIM;
  const float* kn = knew + (size_t)bh * SEQ_T  * HDIM;
  const float* vn = vnew + (size_t)bh * SEQ_T  * HDIM;

  // Q fragments: A 16x32 over HD=128 -> 4 frags (loaded once, reused all chunks)
  v16bf qa[4];
#pragma unroll
  for (int j = 0; j < 4; ++j) qa[j] = load_A_rm(qb + j * 32, HDIM);

  v8f   o[8] = {};                         // O^T accumulator: [d=j*16+mh+r][q=lane&15]
  float mst[8], lst[8];                    // softmax row stats for q rows mh..mh+7
#pragma unroll
  for (int r = 0; r < 8; ++r) { mst[r] = -1e30f; lst[r] = 0.0f; }

  const float scale = 0.08838834764831845f;  // 1/sqrt(128)
  int nn = lane & 15, mh = (lane >> 4) << 3;
  __bf16* pl = &P_lds[wave][0];

  for (int c = wave; c <= NCHUNK; c += 8) {
    bool partial = (c == NCHUNK);          // last chunk: 16 new keys only
    const float* ksrc = partial ? kn : Kb + (size_t)c * 32 * HDIM;
    const float* vsrc = partial ? vn : Vb + (size_t)c * 32 * HDIM;

    // prefetch this wave's next chunk (32 lanes x 512B = 16KB each of K and V)
    if (c + 8 < NCHUNK) {
      const char* pk = (const char*)(Kb + (size_t)(c + 8) * 32 * HDIM) + lane * 512;
      const char* pv = (const char*)(Vb + (size_t)(c + 8) * 32 * HDIM) + lane * 512;
      __builtin_prefetch(pk, 0, 1);
      __builtin_prefetch(pv, 0, 1);
    }

    // S = Q * K_chunk^T  (two 16x16 tiles; C layout: q = mh + r, key = nn)
    v8f s0 = {}, s1 = {};
#pragma unroll
    for (int j = 0; j < 4; ++j) {
      v16bf bk0 = load_BT_rm(ksrc + j * 32, HDIM);                // keys 0..15
      s0 = wmma_bf16<false, false>(qa[j], bk0, s0);
      if (!partial) {
        v16bf bk1 = load_BT_rm(ksrc + 16 * HDIM + j * 32, HDIM);  // keys 16..31
        s1 = wmma_bf16<true, false>(qa[j], bk1, s1);              // same A as prev
      }
    }

    // online softmax: row max across the 16-lane group
    float rmax[8];
#pragma unroll
    for (int r = 0; r < 8; ++r) {
      float v = s0[r] * scale;
      if (!partial) v = fmaxf(v, s1[r] * scale);
      rmax[r] = v;
    }
    for (int off = 1; off < 16; off <<= 1)
#pragma unroll
      for (int r = 0; r < 8; ++r) rmax[r] = fmaxf(rmax[r], __shfl_xor(rmax[r], off, 16));

    float alpha[8], rsum[8];
#pragma unroll
    for (int r = 0; r < 8; ++r) {
      float mnew = fmaxf(mst[r], rmax[r]);
      alpha[r] = __expf(mst[r] - mnew);
      mst[r] = mnew;
      float p0 = __expf(s0[r] * scale - mnew);
      float p1 = partial ? 0.0f : __expf(s1[r] * scale - mnew);
      rsum[r] = p0 + p1;
      int row = mh + r;
      pl[row * 32 + nn]      = (__bf16)p0;
      pl[row * 32 + 16 + nn] = (__bf16)p1;
    }
    // broadcast alpha (row-indexed) to lane-indexed q for the transposed O
    if ((lane & 15) == 0) {
#pragma unroll
      for (int r = 0; r < 8; ++r) a_bcast[wave][mh + r] = alpha[r];
    }
    for (int off = 1; off < 16; off <<= 1)
#pragma unroll
      for (int r = 0; r < 8; ++r) rsum[r] += __shfl_xor(rsum[r], off, 16);
#pragma unroll
    for (int r = 0; r < 8; ++r) lst[r] = lst[r] * alpha[r] + rsum[r];

    // rescale O^T accumulators: q lives in the lane dimension
    float av = a_bcast[wave][nn];          // same-wave LDS, in-order
#pragma unroll
    for (int j = 0; j < 8; ++j)
#pragma unroll
      for (int r = 0; r < 8; ++r) o[j][r] *= av;

    // B-fragment of P^T (32 keys x 16 q): b[i] = P[q=nn][kb+i], b[8+i] = P[q=nn][16+kb+i]
    v16bf pb;
    {
      int kb = (lane >> 4) << 3;
#pragma unroll
      for (int i = 0; i < 8; ++i) pb[i] = pl[nn * 32 + kb + i];
#pragma unroll
      for (int i = 0; i < 8; ++i) pb[8 + i] = pl[nn * 32 + 16 + kb + i];
    }

    // O^T += V^T(16d x 32k) * P^T(32k x 16q), 8 d-tiles; B is invariant -> reuse
#pragma unroll
    for (int j = 0; j < 8; ++j) {
      v16bf va = load_A_vt(vsrc, j * 16, partial);
      if (j == 0) o[j] = wmma_bf16<false, false>(va, pb, o[j]);
      else        o[j] = wmma_bf16<false, true >(va, pb, o[j]);
    }
  }

  // stash this wave's partial (O^T -> [q][d], m, l) and combine across 8 waves
#pragma unroll
  for (int j = 0; j < 8; ++j)
#pragma unroll
    for (int r = 0; r < 8; ++r)
      O_lds[wave][nn * HDIM + j * 16 + mh + r] = o[j][r];
  if ((lane & 15) == 0)
#pragma unroll
    for (int r = 0; r < 8; ++r) { m_lds[wave][mh + r] = mst[r]; l_lds[wave][mh + r] = lst[r]; }
  __syncthreads();

  for (int idx = threadIdx.x; idx < SEQ_T * HDIM; idx += blockDim.x) {
    int tq = idx >> 7, d = idx & 127;
    float M = -1e30f;
#pragma unroll
    for (int w = 0; w < 8; ++w) M = fmaxf(M, m_lds[w][tq]);
    float acc = 0.0f, L = 0.0f;
#pragma unroll
    for (int w = 0; w < 8; ++w) {
      float e = __expf(m_lds[w][tq] - M);
      acc += e * O_lds[w][tq * HDIM + d];
      L   += e * l_lds[w][tq];
    }
    // write pre-projection activations in [B*T, D] layout for the final GEMM
    attn_ws[((size_t)(b * SEQ_T + tq)) * DIM + h * HDIM + d] = acc / L;
  }
}

// ---- kernel 3: output projection (TDM-staged weight tiles) ------------------
__global__ __launch_bounds__(128) void oproj_kernel(
    const float* __restrict__ a_in, const float* __restrict__ Wo,
    const float* __restrict__ bo, float* __restrict__ out) {
#if TDM_OK
  __shared__ __align__(16) float wstage[4][2][WTILE_DWORDS];
#endif
  int wave = threadIdx.x >> 5;
  int lane = threadIdx.x & 31;
  int wid  = blockIdx.x * 4 + wave;        // 256 tasks
  int m0   = (wid >> 5) << 4;
  int n0   = (wid & 31) << 6;

  v8f c[4] = {};
#if TDM_OK
  unsigned lds_off[2] = {
      (unsigned)(unsigned long long)(void*)&wstage[wave][0][0],
      (unsigned)(unsigned long long)(void*)&wstage[wave][1][0]};
  tdm_load_w_tile(Wo + (size_t)n0 * DIM, lds_off[0]);
  for (int k0 = 0; k0 < DIM; k0 += 32) {
    int cur = (k0 >> 5) & 1;
    if (k0 + 32 < DIM) {
      tdm_load_w_tile(Wo + (size_t)n0 * DIM + (k0 + 32), lds_off[cur ^ 1]);
      __builtin_amdgcn_s_wait_tensorcnt(1);
    } else {
      __builtin_amdgcn_s_wait_tensorcnt(0);
    }
    v16bf a = load_A_rm(a_in + (size_t)m0 * DIM + k0, DIM);
#pragma unroll
    for (int j = 0; j < 4; ++j) {
      v16bf b = load_BT_lds(&wstage[wave][cur][0], j);
      if (j == 0) c[j] = wmma_bf16<false, false>(a, b, c[j]);
      else        c[j] = wmma_bf16<true,  false>(a, b, c[j]);
    }
  }
#else
  for (int k0 = 0; k0 < DIM; k0 += 32) {
    v16bf a = load_A_rm(a_in + (size_t)m0 * DIM + k0, DIM);
#pragma unroll
    for (int j = 0; j < 4; ++j) {
      v16bf b = load_BT_rm(Wo + (size_t)(n0 + j * 16) * DIM + k0, DIM);
      if (j == 0) c[j] = wmma_bf16<false, false>(a, b, c[j]);
      else        c[j] = wmma_bf16<true,  false>(a, b, c[j]);
    }
  }
#endif

  int nn = lane & 15, mh = (lane >> 4) << 3;
#pragma unroll
  for (int j = 0; j < 4; ++j)
#pragma unroll
    for (int r = 0; r < 8; ++r) {
      int row = m0 + mh + r, col = n0 + j * 16 + nn;
      out[(size_t)row * DIM + col] = c[j][r] + bo[col];
    }
}

extern "C" void kernel_launch(void* const* d_in, const int* in_sizes, int n_in,
                              void* d_out, int out_size, void* d_ws, size_t ws_size,
                              hipStream_t stream) {
  const float* x  = (const float*)d_in[0];
  const float* Kc = (const float*)d_in[1];
  const float* Vc = (const float*)d_in[2];
  const float* Wq = (const float*)d_in[3];
  const float* bq = (const float*)d_in[4];
  const float* Wk = (const float*)d_in[5];
  const float* bk = (const float*)d_in[6];
  const float* Wv = (const float*)d_in[7];
  const float* bv = (const float*)d_in[8];
  const float* Wo = (const float*)d_in[9];
  const float* bo = (const float*)d_in[10];

  float* out  = (float*)d_out;             // [B,T,D]
  float* kout = out + OUT_ELEMS;           // [B,H,T,HD]
  float* vout = out + 2 * OUT_ELEMS;       // [B,H,T,HD]
  float* qws  = (float*)d_ws;              // [B,H,T,HD] (1 MB)
  float* aws  = qws + OUT_ELEMS;           // [B*T, D]   (1 MB)

  // 768 wave-tasks / 4 waves = 192 blocks
  hipLaunchKernelGGL(qkv_proj_kernel, dim3(192), dim3(128), 0, stream,
                     x, Wq, bq, Wk, bk, Wv, bv, qws, kout, vout);
  // one block per (b,h)
  hipLaunchKernelGGL(attn_kernel, dim3(BATCH * NHEAD), dim3(256), 0, stream,
                     qws, Kc, Vc, kout, vout, aws);
  // 256 wave-tasks / 4 waves = 64 blocks
  hipLaunchKernelGGL(oproj_kernel, dim3(64), dim3(128), 0, stream,
                     aws, Wo, bo, out);
}
